// EquivariantAttention_54202487276125
// MI455X (gfx1250) — compile-verified, matrix operand order checked by reading, not run on hardware
//
#include <hip/hip_runtime.h>

// B=2, N=256, DIM=256, H=4, DH=64, M=16, FF=4, EIN=137, E2=274 (padded to 288)
#define B_   2
#define N_   256
#define DIM_ 256
#define H_   4
#define DH_  64
#define M_   16
#define E2_  274
#define KP_  288

typedef __attribute__((ext_vector_type(16))) _Float16 v16h;
typedef __attribute__((ext_vector_type(8)))  float    v8f;

union H16 { v16h v; _Float16 e[16]; };

// ---------------------------------------------------------------------------
// Prep: We2 (274x16 f32) -> (288x16 f16) zero-padded, and [Wa1|Wc1] (16x64 each)
// -> w1h (32x128 f16) with K rows 16..31 zeroed (so head-GEMM B frags need no
// predication).
__global__ __launch_bounds__(256) void ea_prep(const float* __restrict__ We2,
                                               const float* __restrict__ Wa1,
                                               const float* __restrict__ Wc1,
                                               _Float16* __restrict__ we2h,
                                               _Float16* __restrict__ w1h) {
  int t = blockIdx.x * 256 + threadIdx.x;
  if (t < KP_ * 16) {
    int k = t >> 4;
    we2h[t] = (k < E2_) ? (_Float16)We2[t] : (_Float16)0.f;
  }
  int u = t - KP_ * 16;
  if (u >= 0 && u < 32 * 128) {
    int k = u >> 7, cc = u & 127;
    float v = 0.f;
    if (k < 16) v = (cc < 64) ? Wa1[k * 64 + cc] : Wc1[k * 64 + (cc - 64)];
    w1h[u] = (_Float16)v;
  }
}

// ---------------------------------------------------------------------------
// qkv = feats(512x256) @ Wqkv(256x768); scatter into q/k/v [b][h][n][d] f32.
__global__ __launch_bounds__(32) void ea_qkv(const float* __restrict__ feats,
                                             const float* __restrict__ Wqkv,
                                             float* __restrict__ qbuf,
                                             float* __restrict__ kbuf,
                                             float* __restrict__ vbuf) {
  const int rt = blockIdx.x, ct = blockIdx.y, lane = threadIdx.x;
  const int row = rt * 16 + (lane & 15);
  const int col = ct * 16 + (lane & 15);
  const int kb  = (lane & 16) ? 8 : 0;
  v8f c = {};
  for (int kk = 0; kk < DIM_ / 32; ++kk) {
    H16 a, bb;
#pragma unroll
    for (int t = 0; t < 16; ++t) {
      int K = kk * 32 + ((t < 8) ? (kb + t) : (16 + kb + (t - 8)));
      a.e[t]  = (_Float16)feats[row * DIM_ + K];
      bb.e[t] = (_Float16)Wqkv[K * 768 + col];
    }
    c = __builtin_amdgcn_wmma_f32_16x16x32_f16(false, a.v, false, bb.v,
                                               (short)0, c, false, false);
  }
  const int sect = col >> 8;
  const int h    = (col & 255) >> 6;
  const int d    = col & 63;
  float* dst = (sect == 0) ? qbuf : ((sect == 1) ? kbuf : vbuf);
  const int r0 = (lane & 16) ? 8 : 0;
#pragma unroll
  for (int r = 0; r < 8; ++r) {
    int g = rt * 16 + r0 + r;             // b*N + n
    int b = g >> 8, n = g & 255;
    dst[((b * H_ + h) * N_ + n) * DH_ + d] = c[r];
  }
}

// ---------------------------------------------------------------------------
// Aq/Ak: src(2048x64) @ We1[rowoff:rowoff+64] (64x274) -> out(2048x288), pad=0.
__global__ __launch_bounds__(32) void ea_aqak(const float* __restrict__ src,
                                              const float* __restrict__ We1,
                                              const float* __restrict__ be1, // nullptr for Ak
                                              int rowoff,
                                              float* __restrict__ out) {
  const int rt = blockIdx.x, ct = blockIdx.y, lane = threadIdx.x;
  const int row = rt * 16 + (lane & 15);
  const int col = ct * 16 + (lane & 15);
  const int kb  = (lane & 16) ? 8 : 0;
  v8f c = {};
  for (int kk = 0; kk < 2; ++kk) {
    H16 a, bb;
#pragma unroll
    for (int t = 0; t < 16; ++t) {
      int K = kk * 32 + ((t < 8) ? (kb + t) : (16 + kb + (t - 8)));
      a.e[t]  = (_Float16)src[row * DH_ + K];
      bb.e[t] = (col < E2_) ? (_Float16)We1[(rowoff + K) * E2_ + col] : (_Float16)0.f;
    }
    c = __builtin_amdgcn_wmma_f32_16x16x32_f16(false, a.v, false, bb.v,
                                               (short)0, c, false, false);
  }
  const float bias = (be1 != nullptr && col < E2_) ? be1[col] : 0.f;
  const int r0 = (lane & 16) ? 8 : 0;
#pragma unroll
  for (int r = 0; r < 8; ++r) {
    int g = rt * 16 + r0 + r;
    out[g * KP_ + col] = (col < E2_) ? (c[r] + bias) : 0.f;
  }
}

// ---------------------------------------------------------------------------
// Fused edge pipeline: one block per (b,i); wave w handles head h=w.
__global__ __launch_bounds__(128) void ea_edge(
    const float* __restrict__ coors, const float* __restrict__ basis,
    const float* __restrict__ Aq,    const float* __restrict__ Ak,
    const _Float16* __restrict__ we2h, const float* __restrict__ be2,
    const float* __restrict__ We1,   const _Float16* __restrict__ w1h,
    const float* __restrict__ ba1, const float* __restrict__ Wa2,
    const float* __restrict__ ba2,
    const float* __restrict__ bc1, const float* __restrict__ Wc2,
    const float* __restrict__ bc2,
    const float* __restrict__ vbuf, float* __restrict__ obuf,
    float* __restrict__ coors_out) {
  __shared__ _Float16 sWe2[KP_ * 16];   // 9216 B
  __shared__ _Float16 sW1h[32 * 128];   // 8192 B  [Wa1|Wc1], K rows 16..31 = 0
  __shared__ float sAf[16 * KP_];       // 18432 B
  __shared__ float sAq[H_ * KP_];       // 4608 B
  __shared__ float sFe[16 * 9];
  __shared__ float sSim[H_ * N_];
  __shared__ float sCw[H_ * N_];
  __shared__ float sM[H_ * 16 * 16];
  __shared__ float sW2a[64], sW2c[64], sB1a[64], sB1c[64];
  __shared__ float sCwSum[N_];

  const int tid  = threadIdx.x;
  const int wave = tid >> 5;
  const int lane = tid & 31;
  const int bi   = blockIdx.x;
  const int b    = bi >> 8;
  const int i    = bi & 255;

  for (int t = tid; t < KP_ * 16; t += 128) sWe2[t] = we2h[t];
  for (int t = tid; t < 32 * 128; t += 128) sW1h[t] = w1h[t];
  for (int t = tid; t < H_ * KP_; t += 128) {
    int h = t / KP_, e = t - h * KP_;
    sAq[t] = Aq[((b * H_ + h) * N_ + i) * KP_ + e];
  }
  if (tid < 64) { sW2a[tid] = Wa2[tid]; sW2c[tid] = Wc2[tid];
                  sB1a[tid] = ba1[tid]; sB1c[tid] = bc1[tid]; }
  const float cx = coors[(b * N_ + i) * 3 + 0];
  const float cy = coors[(b * N_ + i) * 3 + 1];
  const float cz = coors[(b * N_ + i) * 3 + 2];
  const float ba2v = ba2[0], bc2v = bc2[0];
  __syncthreads();

  const int h    = wave;                 // head
  const int arow = lane & 15;            // WMMA A row  (j within tile)
  const int kb   = (lane & 16) ? 8 : 0;  // A/B K sub-base per lane half
  const int col  = lane & 15;            // WMMA C col
  const float bcol = be2[col];

  for (int jt = 0; jt < 16; ++jt) {
    const int j0 = jt * 16;
    if (jt < 15)  // prefetch next tile's Ak rows (global_prefetch_b8)
      __builtin_prefetch(Ak + ((b * H_ + h) * N_ + j0 + 16 + arow) * KP_, 0, 1);
    // Fourier features for this j-tile
    if (tid < 16) {
      int jj = j0 + tid;
      float dx = cx - coors[(b * N_ + jj) * 3 + 0];
      float dy = cy - coors[(b * N_ + jj) * 3 + 1];
      float dz = cz - coors[(b * N_ + jj) * 3 + 2];
      float rd = dx * dx + dy * dy + dz * dz;
#pragma unroll
      for (int s = 0; s < 4; ++s) {
        float x = rd * (1.0f / (float)(1 << s));
        sFe[tid * 9 + s]     = __sinf(x);
        sFe[tid * 9 + 4 + s] = __cosf(x);
      }
      sFe[tid * 9 + 8] = rd;
    }
    __syncthreads();
    // Af = fe @ We1[128:137]  (shared across heads)
    for (int t = tid; t < 16 * KP_; t += 128) {
      int j = t / KP_, e = t - j * KP_;
      float acc = 0.f;
      if (e < E2_) {
#pragma unroll
        for (int u = 0; u < 9; ++u) acc += sFe[j * 9 + u] * We1[(128 + u) * E2_ + e];
      }
      sAf[t] = acc;
    }
    __syncthreads();

    // m = relu( relu(Aq+Ak+Af) @ We2 + be2 ) via 9 WMMA k-chunks
    v8f c = {};
    const float* akrow = Ak + ((b * H_ + h) * N_ + j0 + arow) * KP_;
    const float* aqrow = sAq + h * KP_;
    const float* afrow = sAf + arow * KP_;
    for (int kk = 0; kk < KP_ / 32; ++kk) {
      H16 a, bb;
#pragma unroll
      for (int t = 0; t < 16; ++t) {
        int K = kk * 32 + ((t < 8) ? (kb + t) : (16 + kb + (t - 8)));
        float v1 = aqrow[K] + akrow[K] + afrow[K];
        a.e[t]  = (_Float16)fmaxf(v1, 0.f);
        bb.e[t] = sWe2[K * 16 + col];
      }
      c = __builtin_amdgcn_wmma_f32_16x16x32_f16(false, a.v, false, bb.v,
                                                 (short)0, c, false, false);
    }
    {
      const int r0 = (lane & 16) ? 8 : 0;
#pragma unroll
      for (int r = 0; r < 8; ++r)
        sM[h * 256 + (r0 + r) * 16 + col] = fmaxf(c[r] + bcol, 0.f);
    }

    // Head MLPs as WMMA: s1 = relu(m @ [Wa1|Wc1] + b1); out = s1 @ W2 + b2.
    // A = m (16x16, K padded to 32 with zeros); two heads x four 16-col tiles.
    H16 am;
#pragma unroll
    for (int t = 0; t < 16; ++t)
      am.e[t] = (t < 8) ? (_Float16)sM[h * 256 + arow * 16 + (kb + t)]
                        : (_Float16)0.f;

    float simp[8], cwp[8];
#pragma unroll
    for (int r = 0; r < 8; ++r) { simp[r] = 0.f; cwp[r] = 0.f; }
#pragma unroll
    for (int ht = 0; ht < 2; ++ht) {
      const float* b1p = ht ? sB1c : sB1a;
      const float* w2p = ht ? sW2c : sW2a;
      for (int ct = 0; ct < 4; ++ct) {
        H16 bbh;
#pragma unroll
        for (int t = 0; t < 16; ++t) {
          int K = (t < 8) ? (kb + t) : (16 + kb + (t - 8));
          bbh.e[t] = sW1h[K * 128 + ht * 64 + ct * 16 + col];
        }
        v8f zc = {};
        v8f cc = __builtin_amdgcn_wmma_f32_16x16x32_f16(false, am.v, false, bbh.v,
                                                        (short)0, zc, false, false);
        const int tg = ct * 16 + col;
        const float b1v = b1p[tg], w2v = w2p[tg];
#pragma unroll
        for (int r = 0; r < 8; ++r) {
          float hv = fmaxf(cc[r] + b1v, 0.f) * w2v;
          if (ht == 0) simp[r] += hv; else cwp[r] += hv;
        }
      }
    }
    // reduce over the 16 hidden-column lanes of each half (xor<16 stays in half)
#pragma unroll
    for (int r = 0; r < 8; ++r) {
#pragma unroll
      for (int off = 8; off > 0; off >>= 1) {
        simp[r] += __shfl_xor(simp[r], off, 32);
        cwp[r]  += __shfl_xor(cwp[r],  off, 32);
      }
    }
    if ((lane & 15) == 0) {
      const int r0 = (lane & 16) ? 8 : 0;
#pragma unroll
      for (int r = 0; r < 8; ++r) {
        sSim[h * N_ + j0 + r0 + r] = simp[r] + ba2v;
        sCw [h * N_ + j0 + r0 + r] = cwp[r]  + bc2v;
      }
    }
    __syncthreads();
  }

  // softmax over j (per wave == per head), then o = attn @ v
  {
    float mx = -3.0e38f;
    for (int jj = lane; jj < N_; jj += 32) mx = fmaxf(mx, sSim[h * N_ + jj]);
#pragma unroll
    for (int off = 16; off > 0; off >>= 1) mx = fmaxf(mx, __shfl_xor(mx, off, 32));
    float sum = 0.f;
    for (int jj = lane; jj < N_; jj += 32) {
      float ev = __expf(sSim[h * N_ + jj] - mx);
      sSim[h * N_ + jj] = ev;
      sum += ev;
    }
#pragma unroll
    for (int off = 16; off > 0; off >>= 1) sum += __shfl_xor(sum, off, 32);
    const float inv = 1.0f / sum;
    float acc0 = 0.f, acc1 = 0.f;
    for (int jj = 0; jj < N_; ++jj) {
      float aw = sSim[h * N_ + jj] * inv;
      const float* vr = vbuf + ((b * H_ + h) * N_ + jj) * DH_;
      acc0 += aw * vr[lane];
      acc1 += aw * vr[lane + 32];
    }
    obuf[(b * N_ + i) * 256 + h * DH_ + lane]      = acc0;
    obuf[(b * N_ + i) * 256 + h * DH_ + lane + 32] = acc1;
  }
  __syncthreads();
  for (int t = tid; t < N_; t += 128)
    sCwSum[t] = sCw[t] + sCw[N_ + t] + sCw[2 * N_ + t] + sCw[3 * N_ + t];
  __syncthreads();
  if (tid < 3) {
    float acc = coors[(b * N_ + i) * 3 + tid];
    for (int jj = 0; jj < N_; ++jj)
      acc += sCwSum[jj] * basis[((b * N_ + i) * N_ + jj) * 3 + tid];
    coors_out[(b * N_ + i) * 3 + tid] = acc;
  }
}

// ---------------------------------------------------------------------------
// feats_out = obuf(512x256) @ Wout(256x256) + bout
__global__ __launch_bounds__(32) void ea_wout(const float* __restrict__ obuf,
                                              const float* __restrict__ Wout,
                                              const float* __restrict__ bout,
                                              float* __restrict__ out) {
  const int rt = blockIdx.x, ct = blockIdx.y, lane = threadIdx.x;
  const int row = rt * 16 + (lane & 15);
  const int col = ct * 16 + (lane & 15);
  const int kb  = (lane & 16) ? 8 : 0;
  v8f c = {};
  for (int kk = 0; kk < 8; ++kk) {
    H16 a, bb;
#pragma unroll
    for (int t = 0; t < 16; ++t) {
      int K = kk * 32 + ((t < 8) ? (kb + t) : (16 + kb + (t - 8)));
      a.e[t]  = (_Float16)obuf[row * 256 + K];
      bb.e[t] = (_Float16)Wout[K * 256 + col];
    }
    c = __builtin_amdgcn_wmma_f32_16x16x32_f16(false, a.v, false, bb.v,
                                               (short)0, c, false, false);
  }
  const float bias = bout[col];
  const int r0 = (lane & 16) ? 8 : 0;
#pragma unroll
  for (int r = 0; r < 8; ++r) out[(rt * 16 + r0 + r) * 256 + col] = c[r] + bias;
}

// ---------------------------------------------------------------------------
extern "C" void kernel_launch(void* const* d_in, const int* in_sizes, int n_in,
                              void* d_out, int out_size, void* d_ws, size_t ws_size,
                              hipStream_t stream) {
  (void)in_sizes; (void)n_in; (void)out_size; (void)ws_size;
  const float* feats = (const float*)d_in[0];
  const float* coors = (const float*)d_in[1];
  const float* basis = (const float*)d_in[2];
  const float* Wqkv  = (const float*)d_in[3];
  const float* Wout  = (const float*)d_in[4];
  const float* bout  = (const float*)d_in[5];
  const float* We1   = (const float*)d_in[6];
  const float* be1   = (const float*)d_in[7];
  const float* We2   = (const float*)d_in[8];
  const float* be2   = (const float*)d_in[9];
  const float* Wa1   = (const float*)d_in[10];
  const float* ba1   = (const float*)d_in[11];
  const float* Wa2   = (const float*)d_in[12];
  const float* ba2   = (const float*)d_in[13];
  const float* Wc1   = (const float*)d_in[14];
  const float* bc1   = (const float*)d_in[15];
  const float* Wc2   = (const float*)d_in[16];
  const float* bc2   = (const float*)d_in[17];

  float* ws   = (float*)d_ws;
  float* qbuf = ws;                      // 131072
  float* kbuf = qbuf + 131072;           // 131072
  float* vbuf = kbuf + 131072;           // 131072
  float* Aq   = vbuf + 131072;           // 2*4*256*288 = 589824
  float* Ak   = Aq + 589824;             // 589824
  float* obuf = Ak + 589824;             // 131072
  _Float16* we2h = (_Float16*)(obuf + 131072);   // 4608 halves
  _Float16* w1h  = we2h + KP_ * 16;              // 4096 halves

  float* outFeats = (float*)d_out;       // (B,N,DIM) = 131072
  float* outCoors = outFeats + 131072;   // (B,N,3)   = 1536

  ea_prep<<<dim3(34), dim3(256), 0, stream>>>(We2, Wa1, Wc1, we2h, w1h);
  ea_qkv<<<dim3(32, 48), dim3(32), 0, stream>>>(feats, Wqkv, qbuf, kbuf, vbuf);
  ea_aqak<<<dim3(128, 18), dim3(32), 0, stream>>>(qbuf, We1, be1, 0, Aq);
  ea_aqak<<<dim3(128, 18), dim3(32), 0, stream>>>(kbuf, We1, nullptr, 64, Ak);
  ea_edge<<<dim3(512), dim3(128), 0, stream>>>(coors, basis, Aq, Ak, we2h, be2, We1,
                                               w1h, ba1, Wa2, ba2, bc1, Wc2, bc2,
                                               vbuf, obuf, outCoors);
  ea_wout<<<dim3(32, 16), dim3(32), 0, stream>>>(obuf, Wout, bout, outFeats);
}
